// AxonalConnections_76364518523346
// MI455X (gfx1250) — matrix operand order repaired and checked by reading, not compiled, not run on hardware
//
#include <hip/hip_runtime.h>

typedef __bf16 bf16_t;
typedef __attribute__((ext_vector_type(16))) __bf16 v16bf;
typedef __attribute__((ext_vector_type(8)))  __bf16 v8bf;
typedef __attribute__((ext_vector_type(8)))  float  v8f;
typedef __attribute__((ext_vector_type(4)))  float  v4f;
typedef __attribute__((ext_vector_type(4)))  unsigned int v4u;
typedef __attribute__((ext_vector_type(8)))  int    v8i;
typedef __attribute__((ext_vector_type(4)))  int    v4i;

#define BM    128   // block tile rows (batch)
#define BN    64    // block tile cols (output features)
#define KT    32    // k-tile (bf16 WMMA K)
#define LDSAF 36    // fp32 LDS row stride: 32 data + 4 pad DWORDs (from TDM pad engine)
#define ROWBF (2 * LDSAF)   // same row in bf16 units after in-place conversion (72)

union FragBF { v16bf v; v8bf h[2]; };
union FragF  { v8f   v; float f[8]; };

// ---- Tensor Data Mover: 2D fp32 tile load, global -> LDS (D# per CDNA5 ISA) ----
static __device__ __forceinline__ void tdm_load_tile_2d(unsigned long long gaddr,
                                                        unsigned int lds_addr,
                                                        unsigned int tile_rows,
                                                        unsigned int row_stride_elems)
{
    const unsigned int TD = 1u << 20;   // generous tensor dims: tile always in-bounds, no clipping
    v4u g0 = {
        1u,                                                   // count=1 user descriptor
        lds_addr,                                             // LDS byte address
        (unsigned int)gaddr,                                  // global_addr[31:0]
        (((unsigned int)(gaddr >> 32)) & 0x01FFFFFFu) | 0x80000000u  // addr[56:32] | type=2
    };
    v8i g1 = {
        (int)((2u << 16) |      // data_size = 4 bytes
              (1u << 20) |      // pad_enable
              (4u << 22) |      // pad_interval code 4 -> every 32 DWORDs
              (3u << 25)),      // pad_amount  code 3 -> 4 DWORDs  (row stride = 36 DWORDs)
        (int)((TD & 0xFFFFu) << 16),                          // tensor_dim0 lo16 (barrier addr = 0)
        (int)(((TD >> 16) & 0xFFFFu) | ((TD & 0xFFFFu) << 16)),// tensor_dim0 hi16 | tensor_dim1 lo16
        (int)(((TD >> 16) & 0xFFFFu) | ((unsigned)KT << 16)), // tensor_dim1 hi16 | tile_dim0 = 32
        (int)(tile_rows & 0xFFFFu),                           // tile_dim1 = rows, tile_dim2 = 0
        (int)row_stride_elems,                                // tensor_dim0_stride lo32 (elements)
        0,                                                    // stride hi16 | dim1_stride lo16
        0
    };
    v4i g2 = { 0, 0, 0, 0 };
    v4i g3 = { 0, 0, 0, 0 };
#if defined(__clang_major__) && __clang_major__ >= 23
    v8i g4 = { 0, 0, 0, 0, 0, 0, 0, 0 };
    __builtin_amdgcn_tensor_load_to_lds(g0, g1, g2, g3, g4, 0);
#else
    __builtin_amdgcn_tensor_load_to_lds(g0, g1, g2, g3, 0);
#endif
}

// convert 8 fp32 (two v4f) into bf16 hi + bf16 residual lo
static __device__ __forceinline__ void cvt8(v4f a, v4f b, v8bf& ho, v8bf& lo)
{
    v8bf hv, lv;
    #pragma unroll
    for (int j = 0; j < 4; ++j) {
        { float f = a[j]; bf16_t hb = (bf16_t)f; hv[j]     = hb; lv[j]     = (bf16_t)(f - (float)hb); }
        { float f = b[j]; bf16_t hb = (bf16_t)f; hv[4 + j] = hb; lv[4 + j] = (bf16_t)(f - (float)hb); }
    }
    ho = hv; lo = lv;
}

__global__ __launch_bounds__(256, 1)
void axonal_wmma_tdm_kernel(const float* __restrict__ x,
                            const float* __restrict__ W,
                            const float* __restrict__ lfreq,
                            const float* __restrict__ gfreq,
                            const float* __restrict__ strength,
                            const int*   __restrict__ clk,
                            float* __restrict__ out,
                            int Bdim, int Ddim)
{
    // double-buffered staging tiles; TDM writes fp32 (36-DWORD rows), then converted
    // in place to [32 x bf16 hi | 32 x bf16 lo | pad] per row.
    __shared__ float stage[(2 * BM + 2 * BN) * LDSAF];

    const int tid  = threadIdx.x;
    const int lane = tid & 31;
    const int wave = tid >> 5;       // 0..7
    const int wm   = wave >> 1;      // 0..3 : 32-row slice of BM
    const int wn   = wave & 1;       // 0..1 : 32-col slice of BN
    const int sel  = lane >> 4;
    const int l15  = lane & 15;

    const int d  = blockIdx.z;
    const int m0 = blockIdx.y * BM;
    const int n0 = blockIdx.x * BN;

    const unsigned int ldsBase = (unsigned int)(uintptr_t)(void*)stage;

    // the 3 (source, pair) feeds for destination d, in PAIRS order
    int slist[3], plist[3];
    {
        int c = 0;
        #pragma unroll
        for (int s = 0; s < 4; ++s)
            if (s != d) { slist[c] = s; plist[c] = s * 3 + d - (d > s ? 1 : 0); ++c; }
    }

    const int ksteps = Ddim / KT;
    const int nsteps = 3 * ksteps;

    // only wave 0 drives the TDM (scalar branch; TDM ignores EXEC)
    const bool leader = (__builtin_amdgcn_readfirstlane((int)tid) >> 5) == 0;

    auto issue = [&](int step, int buf) {
        const int si = step / ksteps;
        const int k0 = (step - si * ksteps) * KT;
        const float* xa = x + ((size_t)slist[si] * Bdim + m0) * Ddim + k0;
        const float* wb = W + ((size_t)plist[si] * Ddim + n0) * (size_t)Ddim + k0;
        tdm_load_tile_2d((unsigned long long)(uintptr_t)xa,
                         ldsBase + (unsigned)(buf * BM * LDSAF) * 4u, BM, (unsigned)Ddim);
        tdm_load_tile_2d((unsigned long long)(uintptr_t)wb,
                         ldsBase + (unsigned)((2 * BM + buf * BN) * LDSAF) * 4u, BN, (unsigned)Ddim);
    };

    FragF ch[2][2], cl[2][2];   // hi*hi accumulator and cross-term accumulator
    #pragma unroll
    for (int mi = 0; mi < 2; ++mi)
        #pragma unroll
        for (int ni = 0; ni < 2; ++ni)
            #pragma unroll
            for (int q = 0; q < 8; ++q) {
                ch[mi][ni].f[q] = 0.0f;
                cl[mi][ni].f[q] = 0.0f;
            }

    // conversion-pass work assignment (once per element)
    const int arow = tid >> 1, ahalf = tid & 1;   // A: half a 32-elem row per thread
    const int brow = tid >> 2, bq    = tid & 3;   // B: quarter row per thread

    if (leader) issue(0, 0);    // prime buffer 0

    for (int step = 0; step < nsteps; ++step) {
        const int cur = step & 1;

        if (leader) {
            if (step + 1 < nsteps) {
                issue(step + 1, cur ^ 1);                 // next tile into the other buffer
                __builtin_amdgcn_s_wait_tensorcnt(2);     // in-order: current buffer's 2 loads done
            } else {
                __builtin_amdgcn_s_wait_tensorcnt(0);
            }
        }
        __syncthreads();   // TDM data for buf[cur] visible to all waves

        float*  sAf = stage + cur * BM * LDSAF;
        float*  sBf = stage + (2 * BM + cur * BN) * LDSAF;
        bf16_t* sAb = (bf16_t*)sAf;
        bf16_t* sBb = (bf16_t*)sBf;

        // ---- phase 1: read fp32 slices into registers ----
        v4f ar0, ar1, ar2, ar3, br0, br1;
        {
            const float* pa = sAf + arow * LDSAF + ahalf * 16;
            ar0 = ((const v4f*)pa)[0];
            ar1 = ((const v4f*)pa)[1];
            ar2 = ((const v4f*)pa)[2];
            ar3 = ((const v4f*)pa)[3];
            const float* pb = sBf + brow * LDSAF + bq * 8;
            br0 = ((const v4f*)pb)[0];
            br1 = ((const v4f*)pb)[1];
        }
        __syncthreads();   // all reads done before in-place overwrite

        // ---- phase 2: in-place bf16 hi/lo split (each element converted exactly once) ----
        {
            v8bf h0, l0, h1, l1;
            cvt8(ar0, ar1, h0, l0);      // A elems k = ahalf*16 .. +8
            cvt8(ar2, ar3, h1, l1);      // A elems k = ahalf*16+8 .. +16
            const int abase = arow * ROWBF + ahalf * 16;
            *(v8bf*)&sAb[abase]          = h0;   // hi region: row elems [0,32)
            *(v8bf*)&sAb[abase + 8]      = h1;
            *(v8bf*)&sAb[abase + 32]     = l0;   // lo region: row elems [32,64)
            *(v8bf*)&sAb[abase + 40]     = l1;

            v8bf hb, lb;
            cvt8(br0, br1, hb, lb);      // B elems k = bq*8 .. +8
            const int bbase = brow * ROWBF + bq * 8;
            *(v8bf*)&sBb[bbase]          = hb;
            *(v8bf*)&sBb[bbase + 32]     = lb;
        }
        __syncthreads();   // converted tile visible

        // ---- phase 3: gather fragments (CDNA5 VGPR layouts) + WMMA ----
        FragBF ahi[2], alo[2], bhi[2], blo[2];
        #pragma unroll
        for (int mi = 0; mi < 2; ++mi) {
            // A 16x32: lane m = l15; lanes<16 hold K 0-7 & 16-23, lanes>=16 hold K 8-15 & 24-31
            const int base = (wm * 32 + mi * 16 + l15) * ROWBF + sel * 8;
            ahi[mi].h[0] = *(const v8bf*)&sAb[base];
            ahi[mi].h[1] = *(const v8bf*)&sAb[base + 16];
            alo[mi].h[0] = *(const v8bf*)&sAb[base + 32];
            alo[mi].h[1] = *(const v8bf*)&sAb[base + 48];
        }
        #pragma unroll
        for (int ni = 0; ni < 2; ++ni) {
            // B 32x16: lane n = l15; lanes<16 hold K 0-7 & 16-23, lanes>=16 hold K 8-15 & 24-31
            const int base = (wn * 32 + ni * 16 + l15) * ROWBF + sel * 8;
            bhi[ni].h[0] = *(const v8bf*)&sBb[base];
            bhi[ni].h[1] = *(const v8bf*)&sBb[base + 16];
            blo[ni].h[0] = *(const v8bf*)&sBb[base + 32];
            blo[ni].h[1] = *(const v8bf*)&sBb[base + 48];
        }

        // split-precision MAC on independent chains: ch += hi*hi ; cl += hi*lo + lo*hi
        #pragma unroll
        for (int mi = 0; mi < 2; ++mi)
            #pragma unroll
            for (int ni = 0; ni < 2; ++ni) {
                ch[mi][ni].v = __builtin_amdgcn_wmma_f32_16x16x32_bf16(
                    false, ahi[mi].v, false, bhi[ni].v, (short)0, ch[mi][ni].v, false, false);
                cl[mi][ni].v = __builtin_amdgcn_wmma_f32_16x16x32_bf16(
                    false, ahi[mi].v, false, blo[ni].v, (short)0, cl[mi][ni].v, false, false);
                cl[mi][ni].v = __builtin_amdgcn_wmma_f32_16x16x32_bf16(
                    false, alo[mi].v, false, bhi[ni].v, (short)0, cl[mi][ni].v, false, false);
            }

        __syncthreads();   // all waves done with buf[cur] before TDM re-fills it
    }

    // ---- epilogue: merge accumulators + fused interference term + writeback ----
    const float t   = 2.0f * 3.14159265358979f * (float)(*clk) * 0.001f;
    const float g   = __sinf(t * gfreq[d]);
    const float str = strength[d];

    #pragma unroll
    for (int ni = 0; ni < 2; ++ni) {
        const int colg = n0 + wn * 32 + ni * 16 + l15;
        const float addc = str * (__sinf(t * lfreq[(size_t)d * Ddim + colg]) + g);
        #pragma unroll
        for (int mi = 0; mi < 2; ++mi) {
            const int rbase = m0 + wm * 32 + mi * 16 + sel * 8;   // C layout: M = v + 8*sel
            #pragma unroll
            for (int v = 0; v < 8; ++v) {
                const int rowg = rbase + v;
                out[((size_t)d * Bdim + rowg) * Ddim + colg] =
                    ch[mi][ni].f[v] + cl[mi][ni].f[v] + addc;
            }
        }
    }
}

extern "C" void kernel_launch(void* const* d_in, const int* in_sizes, int n_in,
                              void* d_out, int out_size, void* d_ws, size_t ws_size,
                              hipStream_t stream) {
    const float* x   = (const float*)d_in[0];   // [4, B, D]
    const float* W   = (const float*)d_in[1];   // [12, D, D]
    const float* lf  = (const float*)d_in[2];   // [4, D]
    const float* gf  = (const float*)d_in[3];   // [4]
    const float* st  = (const float*)d_in[4];   // [4]
    const int*   clk = (const int*)d_in[5];     // scalar

    float* out = (float*)d_out;                 // [4, B, D]

    const int D = in_sizes[2] / 4;              // local_freq is [4, D]
    const int B = in_sizes[0] / (4 * D);        // x is [4, B, D]

    dim3 grid(D / BN, B / BM, 4);
    dim3 block(256, 1, 1);
    axonal_wmma_tdm_kernel<<<grid, block, 0, stream>>>(x, W, lf, gf, st, clk, out, B, D);
}
